// MetaGRUCell_47974784696682
// MI455X (gfx1250) — matrix operand order repaired and checked by reading, not compiled, not run on hardware
//
#include <hip/hip_runtime.h>

// ---------------------------------------------------------------------------
// MetaGRU for MI455X (gfx1250): one workgroup per node.
//  * Hypernetwork evaluated in-kernel (one thread per hyper-MLP trunk),
//    generated weights stored to LDS as f16.
//  * GRU recurrence: all matmuls via v_wmma_f32_16x16x32_f16 (fp32 accum).
//  * h state lives in LDS across all T steps; HBM traffic = stream x, write h.
//  * Activations use v_rcp_f32 (no IEEE division chains).
// ---------------------------------------------------------------------------

#define F_IN 32
#define HID  64
#define FEAT 32
#define B_   64
#define T_   24
#define N_   1024

typedef __attribute__((ext_vector_type(16))) _Float16 v16h;
typedef __attribute__((ext_vector_type(8)))  _Float16 v8h;
typedef __attribute__((ext_vector_type(8)))  float    v8f;
typedef __attribute__((ext_vector_type(4)))  float    v4f;

// LDS strides (elements). Row strides are multiples of 8 halves (16B) so
// per-lane b128 loads stay aligned; odd multiples of 16B spread banks.
#define WZR_S 136   // 128 cols + pad (f16)
#define WI_S  72    // 64 cols + pad  (f16)
#define WH_S  72
#define DS_S  104   // 96 cols + pad  (f16)
#define RH_S  72
#define H_S   65    // 64 cols + pad  (f32)
#define Z_S   65

// LDS byte offsets
#define OFF_WZR 0
#define OFF_WI  (OFF_WZR + 96 * WZR_S * 2)         // 26112
#define OFF_WH  (OFF_WI  + 32 * WI_S  * 2)         // 30720
#define OFF_DS  (OFF_WH  + 64 * WH_S  * 2)         // 39936
#define OFF_RH  (OFF_DS  + 64 * DS_S  * 2)         // 53248
#define OFF_H   (OFF_RH  + 64 * RH_S  * 2)         // 62464
#define OFF_Z   (OFF_H   + 64 * H_S   * 4)         // 79104
#define OFF_BZR (OFF_Z   + 64 * Z_S   * 4)         // 95744
#define OFF_BI  (OFF_BZR + 128 * 4)                // 96256
#define OFF_BH  (OFF_BI  + 64 * 4)                 // 96512
#define OFF_HC  (OFF_BH  + 64 * 4)                 // 96768: 12 hypernet coeffs
#define SMEM_BYTES (OFF_HC + 16 * 4)               // 96832

struct Params {
  const float* feature;   // [N, FEAT]
  const float* data;      // [B, T, N, F_IN]
  // six hyper-MLPs x {W1[32x16], b1[16], W2[16x2], b2[2], W3[2xf], b3[f]}
  // order: zr_w, zr_b, i2h_w, i2h_b, h2h_w, h2h_b
  const float* p[36];
  float* out;             // [B, T, N, HID]
  float* state;           // [B, N, HID]
};

__device__ __forceinline__ float rcp_fast(float x) {
  return __builtin_amdgcn_rcpf(x);              // v_rcp_f32
}
__device__ __forceinline__ float sigmoidf_(float x) {
  return rcp_fast(1.0f + __expf(-x));
}
__device__ __forceinline__ float tanhf_(float x) {
  // 1 - 2/(e^{2x}+1); saturates correctly at +/-1
  return 1.0f - 2.0f * rcp_fast(__expf(2.0f * x) + 1.0f);
}

__device__ __forceinline__ v16h cat16(v8h lo, v8h hi) {
  return __builtin_shufflevector(lo, hi, 0,1,2,3,4,5,6,7,8,9,10,11,12,13,14,15);
}

// A-fragment (16x32 f16, ISA 7.12.2): lane<16 -> row=lane, K {0..7,16..23};
// lane>=16 -> row=lane-16, K {8..15,24..31}.
__device__ __forceinline__ v16h loadA(const _Float16* base, int stride,
                                      int mtile, int ktile, int lane) {
  const _Float16* row = base + (mtile * 16 + (lane & 15)) * stride
                             + ktile * 32 + ((lane >> 4) << 3);
  v8h lo = *(const v8h*)row;
  v8h hi = *(const v8h*)(row + 16);
  return cat16(lo, hi);
}

// B-fragment (32x16 f16): lane<16 -> col=lane, K {0..7,16..23};
// lane>=16 -> col=lane-16, K {8..15,24..31}.
__device__ __forceinline__ v16h loadB(const _Float16* base, int stride,
                                      int ktile, int ntile, int lane) {
  const _Float16* col = base + (ktile * 32 + ((lane >> 4) << 3)) * stride
                             + ntile * 16 + (lane & 15);
  v16h b;
#pragma unroll
  for (int s = 0; s < 8; ++s) {
    b[s]     = col[s * stride];
    b[8 + s] = col[(16 + s) * stride];
  }
  return b;
}

__device__ __forceinline__ v8f wmma(v16h a, v16h b, v8f c) {
  return __builtin_amdgcn_wmma_f32_16x16x32_f16(
      /*neg_a=*/false, a, /*neg_b=*/false, b,
      /*c_mod=*/(short)0, c, /*reuse_a=*/false, /*reuse_b=*/false);
}

// Hypernet trunk: sigmoid(sigmoid(feat@W1+b1)@W2+b2) -> 2 scalars.
__device__ void mlp2(const float* feat, const float* W1, const float* b1,
                     const float* W2, const float* b2, float& o0, float& o1) {
  float a = b2[0], b = b2[1];
#pragma unroll 1
  for (int j = 0; j < 16; ++j) {
    float s = b1[j];
#pragma unroll
    for (int i = 0; i < 32; ++i) s += feat[i] * W1[i * 16 + j];
    const float h1 = sigmoidf_(s);
    a += h1 * W2[j * 2];
    b += h1 * W2[j * 2 + 1];
  }
  o0 = sigmoidf_(a);
  o1 = sigmoidf_(b);
}

__global__ __launch_bounds__(256)
void metagru_kernel(Params P) {
  extern __shared__ __align__(16) char smem[];
  _Float16* wzr = (_Float16*)(smem + OFF_WZR);
  _Float16* wi  = (_Float16*)(smem + OFF_WI);
  _Float16* wh  = (_Float16*)(smem + OFF_WH);
  _Float16* ds  = (_Float16*)(smem + OFF_DS);   // [64 x 96] = [x_t | h] f16
  _Float16* rh  = (_Float16*)(smem + OFF_RH);   // [64 x 64] r*h f16
  float*    hbf = (float*)(smem + OFF_H);       // [64 x 64] h fp32
  float*    zbf = (float*)(smem + OFF_Z);       // [64 x 64] z fp32
  float*    bzr = (float*)(smem + OFF_BZR);     // [128]
  float*    bi  = (float*)(smem + OFF_BI);      // [64]
  float*    bh  = (float*)(smem + OFF_BH);      // [64]
  float*    hc  = (float*)(smem + OFF_HC);      // [12] hypernet coeffs

  const int node  = blockIdx.x;
  const int tid   = threadIdx.x;
  const int lane  = tid & 31;
  const int wv    = tid >> 5;     // 8 waves
  const int mtile = wv & 3;       // batch-tile of this wave
  const int g     = wv >> 2;      // column group

  // ---- hypernet trunks: one thread per hyper-MLP, publish coeffs via LDS ----
  if (tid < 6) {
    const float* feat = P.feature + node * FEAT;
    float o0, o1;
    mlp2(feat, P.p[6 * tid], P.p[6 * tid + 1], P.p[6 * tid + 2], P.p[6 * tid + 3],
         o0, o1);
    hc[2 * tid]     = o0;
    hc[2 * tid + 1] = o1;
  }
  __syncthreads();

  const float hz0  = hc[0],  hz1  = hc[1];    // zr_w
  const float hzb0 = hc[2],  hzb1 = hc[3];    // zr_b
  const float hi0  = hc[4],  hi1  = hc[5];    // i2h_w
  const float hib0 = hc[6],  hib1 = hc[7];    // i2h_b
  const float hh0  = hc[8],  hh1  = hc[9];    // h2h_w
  const float hhb0 = hc[10], hhb1 = hc[11];   // h2h_b

  // ---- generate this node's weights into LDS (f16) ----
  { // Wzr [96 x 128]
    const float* W3 = P.p[4]; const float* b3 = P.p[5];
    for (int j = tid; j < 96 * 128; j += 256) {
      float w = hz0 * W3[j] + hz1 * W3[96 * 128 + j] + b3[j];
      wzr[(j >> 7) * WZR_S + (j & 127)] = (_Float16)w;
    }
  }
  { // Wi [32 x 64]
    const float* W3 = P.p[16]; const float* b3 = P.p[17];
    for (int j = tid; j < 32 * 64; j += 256) {
      float w = hi0 * W3[j] + hi1 * W3[32 * 64 + j] + b3[j];
      wi[(j >> 6) * WI_S + (j & 63)] = (_Float16)w;
    }
  }
  { // Wh [64 x 64]
    const float* W3 = P.p[28]; const float* b3 = P.p[29];
    for (int j = tid; j < 64 * 64; j += 256) {
      float w = hh0 * W3[j] + hh1 * W3[64 * 64 + j] + b3[j];
      wh[(j >> 6) * WH_S + (j & 63)] = (_Float16)w;
    }
  }
  { // biases (fp32) + zero h
    const float* W3z = P.p[10]; const float* b3z = P.p[11];
    if (tid < 128) bzr[tid] = hzb0 * W3z[tid] + hzb1 * W3z[128 + tid] + b3z[tid];
    const float* W3i = P.p[22]; const float* b3i = P.p[23];
    if (tid < 64)  bi[tid] = hib0 * W3i[tid] + hib1 * W3i[64 + tid] + b3i[tid];
    const float* W3h = P.p[34]; const float* b3h = P.p[35];
    if (tid < 64)  bh[tid] = hhb0 * W3h[tid] + hhb1 * W3h[64 + tid] + b3h[tid];
    for (int j = tid; j < B_ * HID; j += 256)
      hbf[(j >> 6) * H_S + (j & 63)] = 0.0f;
  }
  __syncthreads();

  // ---------------- GRU recurrence ----------------
  const int r = tid >> 2;   // batch row this thread stages (0..63)
  const int q = tid & 3;    // quarter within the row

  for (int t = 0; t < T_; ++t) {
    // phase 1: build ds = [x_t | h] in f16
    {
      const float* xp = P.data + ((size_t)(r * T_ + t) * N_ + node) * F_IN + q * 8;
      v4f x0 = *(const v4f*)xp;
      v4f x1 = *(const v4f*)(xp + 4);
      _Float16* d = ds + r * DS_S + q * 8;
#pragma unroll
      for (int e = 0; e < 4; ++e) { d[e] = (_Float16)x0[e]; d[4 + e] = (_Float16)x1[e]; }
      if (t + 1 < T_) __builtin_prefetch(xp + (size_t)N_ * F_IN, 0, 0);

      const float* hrow = hbf + r * H_S + q * 16;
      _Float16*    drow = ds + r * DS_S + 32 + q * 16;
#pragma unroll
      for (int e = 0; e < 16; ++e) drow[e] = (_Float16)hrow[e];
    }
    __syncthreads();

    // phase 2: zr = sigmoid(ds @ Wzr + bzr); z -> zbf, r*h -> rh (f16)
    {
      v16h a0 = loadA(ds, DS_S, mtile, 0, lane);
      v16h a1 = loadA(ds, DS_S, mtile, 1, lane);
      v16h a2 = loadA(ds, DS_S, mtile, 2, lane);
      const int col_in = lane & 15;
      const int rbase  = mtile * 16 + ((lane >> 4) << 3);
#pragma unroll
      for (int nt = 0; nt < 4; ++nt) {
        const int ntile = g * 4 + nt;
        v8f c = {};
        c = wmma(a0, loadB(wzr, WZR_S, 0, ntile, lane), c);
        c = wmma(a1, loadB(wzr, WZR_S, 1, ntile, lane), c);
        c = wmma(a2, loadB(wzr, WZR_S, 2, ntile, lane), c);
        const int col  = ntile * 16 + col_in;   // 0..127
        const float bb = bzr[col];
#pragma unroll
        for (int gp = 0; gp < 8; ++gp) {
          const int m = rbase + gp;
          const float v = sigmoidf_(c[gp] + bb);
          if (g == 0) {
            zbf[m * Z_S + col] = v;                                   // z
          } else {
            const int ch = col - 64;                                  // r
            rh[m * RH_S + ch] = (_Float16)(v * hbf[m * H_S + ch]);
          }
        }
      }
    }
    __syncthreads();

    // phase 3: cand = tanh(x@Wi + (r*h)@Wh + bi + bh); h = z*h + (1-z)*cand
    {
      v16h ax  = loadA(ds, DS_S, mtile, 0, lane);   // x = ds cols 0..31
      v16h ar0 = loadA(rh, RH_S, mtile, 0, lane);
      v16h ar1 = loadA(rh, RH_S, mtile, 1, lane);
      const int col_in = lane & 15;
      const int rbase  = mtile * 16 + ((lane >> 4) << 3);
#pragma unroll
      for (int nt = 0; nt < 2; ++nt) {
        const int ntile = g * 2 + nt;
        v8f c = {};
        c = wmma(ax,  loadB(wi, WI_S, 0, ntile, lane), c);
        c = wmma(ar0, loadB(wh, WH_S, 0, ntile, lane), c);
        c = wmma(ar1, loadB(wh, WH_S, 1, ntile, lane), c);
        const int col  = ntile * 16 + col_in;       // 0..63
        const float bb = bi[col] + bh[col];
#pragma unroll
        for (int gp = 0; gp < 8; ++gp) {
          const int m   = rbase + gp;
          const float cand = tanhf_(c[gp] + bb);
          const float zz = zbf[m * Z_S + col];
          const float hv = hbf[m * H_S + col];
          const float hn = zz * hv + (1.0f - zz) * cand;
          hbf[m * H_S + col] = hn;
          P.out[((size_t)(m * T_ + t) * N_ + node) * HID + col] = hn;
        }
      }
    }
    __syncthreads();
  }

  // final state [B, N, HID]
  for (int j = tid; j < B_ * HID; j += 256) {
    P.state[((size_t)(j >> 6) * N_ + node) * HID + (j & 63)] =
        hbf[(j >> 6) * H_S + (j & 63)];
  }
}

extern "C" void kernel_launch(void* const* d_in, const int* in_sizes, int n_in,
                              void* d_out, int out_size, void* d_ws, size_t ws_size,
                              hipStream_t stream) {
  (void)in_sizes; (void)n_in; (void)d_ws; (void)ws_size; (void)out_size;
  Params P;
  P.feature = (const float*)d_in[0];
  P.data    = (const float*)d_in[1];
  for (int i = 0; i < 36; ++i) P.p[i] = (const float*)d_in[2 + i];
  P.out   = (float*)d_out;
  P.state = (float*)d_out + (size_t)B_ * T_ * N_ * HID;

  dim3 grid(N_), block(256);
  hipLaunchKernelGGL(metagru_kernel, grid, block, SMEM_BYTES, stream, P);
}
